// MockMoELayer_12292196401257
// MI455X (gfx1250) — matrix operand order repaired and checked by reading, not compiled, not run on hardware
//
#include <hip/hip_runtime.h>

// ---------------------------------------------------------------------------
// MoE layer for MI455X (gfx1250, wave32, WMMA).
//   tokens T=2048, hidden H=1024, experts E=8, top-2 (weights NOT applied).
//   out[t] = sum_{e in top2(logits)} (W_e x_t + b_e)
// Strategy: bf16-convert X and W once; route; group tokens by expert;
// dense bf16 WMMA GEMM per expert tile; atomic-add scatter into f32 output.
// ---------------------------------------------------------------------------

#define T_TOK 2048
#define H_DIM 1024
#define N_EXP 8

#define TM 64    // block tile: tokens
#define TN 128   // block tile: output features
#define TK 32    // K step (bf16 WMMA K)

typedef __attribute__((ext_vector_type(16))) __bf16 v16bf;
typedef __attribute__((ext_vector_type(8)))  __bf16 v8bf;
typedef __attribute__((ext_vector_type(8)))  float  v8f;

union V16u {
    v16bf v;
    struct { v8bf lo, hi; } h;
};

static __device__ __forceinline__ unsigned short f2bf(float f) {
    union { float f; unsigned int u; } c; c.f = f;
    unsigned int u = c.u;
    unsigned int r = u + 0x7FFFu + ((u >> 16) & 1u);   // round-to-nearest-even
    return (unsigned short)(r >> 16);
}

static __device__ __forceinline__ int imin(int a, int b) { return a < b ? a : b; }

// ---------------------------------------------------------------- zero counts
__global__ void zero_counts_kernel(int* counts) {
    if (threadIdx.x < N_EXP) counts[threadIdx.x] = 0;
}

// ------------------------------------------------------------- f32 -> bf16
__global__ void cvt_bf16_kernel(const float* __restrict__ src,
                                unsigned short* __restrict__ dst, int n) {
    int i = (blockIdx.x * blockDim.x + threadIdx.x) * 4;
    int stride = gridDim.x * blockDim.x * 4;
    for (; i < n; i += stride) {
        float4 f = *(const float4*)(src + i);
        ushort4 o;
        o.x = f2bf(f.x); o.y = f2bf(f.y); o.z = f2bf(f.z); o.w = f2bf(f.w);
        *(ushort4*)(dst + i) = o;
    }
}

// ------------------------------------------------------------------- router
// One wave per token: 8 lane-parallel dot products, butterfly reduce,
// top-2 argmax of logits (softmax is monotonic; weights unused in reference).
__global__ __launch_bounds__(256)
void router_kernel(const float* __restrict__ hs,
                   const float* __restrict__ rw,
                   const float* __restrict__ rb,
                   int* __restrict__ counts,
                   int* __restrict__ lists,
                   int* __restrict__ top2) {
    int lane = threadIdx.x & 31;
    int wave = threadIdx.x >> 5;
    int t = blockIdx.x * 8 + wave;

    float acc[N_EXP];
#pragma unroll
    for (int e = 0; e < N_EXP; ++e) acc[e] = 0.f;

    const float* x = hs + (size_t)t * H_DIM;
    for (int i = 0; i < H_DIM / 32; ++i) {
        int h = i * 32 + lane;
        float xv = x[h];
#pragma unroll
        for (int e = 0; e < N_EXP; ++e) acc[e] += xv * rw[e * H_DIM + h];
    }
#pragma unroll
    for (int e = 0; e < N_EXP; ++e) {
#pragma unroll
        for (int off = 16; off; off >>= 1)
            acc[e] += __shfl_xor(acc[e], off, 32);
        acc[e] += rb[e];
    }
    // top-2, first occurrence wins ties (matches jax.lax.top_k)
    int e0 = 0; float m0 = acc[0];
#pragma unroll
    for (int e = 1; e < N_EXP; ++e) if (acc[e] > m0) { m0 = acc[e]; e0 = e; }
    int e1 = -1; float m1 = -3.4e38f;
#pragma unroll
    for (int e = 0; e < N_EXP; ++e)
        if (e != e0 && acc[e] > m1) { m1 = acc[e]; e1 = e; }

    if (lane == 0) {
        int p0 = atomicAdd(&counts[e0], 1);
        lists[e0 * T_TOK + p0] = t;
        int p1 = atomicAdd(&counts[e1], 1);
        lists[e1 * T_TOK + p1] = t;
        top2[t] = e0 | (e1 << 8);
    }
}

// ------------------------------------------------------- bias initialization
// Fully (re)writes d_out each call: out[t] = b_{e0} + b_{e1}.
__global__ void bias_init_kernel(const float* __restrict__ eb,
                                 const int* __restrict__ top2,
                                 float* __restrict__ out) {
    int t = blockIdx.x;
    int h = threadIdx.x * 4;
    int pk = top2[t];
    int e0 = pk & 0xFF, e1 = (pk >> 8) & 0xFF;
    float4 b0 = *(const float4*)(eb + e0 * H_DIM + h);
    float4 b1 = *(const float4*)(eb + e1 * H_DIM + h);
    float4 o;
    o.x = b0.x + b1.x; o.y = b0.y + b1.y;
    o.z = b0.z + b1.z; o.w = b0.w + b1.w;
    *(float4*)(out + (size_t)t * H_DIM + h) = o;
}

// ----------------------------------------------------------- expert GEMM
// Block tile 64(M tokens) x 128(N features), K-step 32. 8 waves, each wave
// owns a 32x32 sub-tile = 2x2 v_wmma_f32_16x16x32_bf16 accumulators.
// A fragment (16x32 bf16): lane L holds row M=L&15; low lanes K{0..15},
// high lanes K{16..31}. B fragment (32x16): lane L holds col N=L&15,
// K = (L>>4)*16 .. +15 -> one contiguous 32B slice of a W_e row.
__global__ __launch_bounds__(256)
void moe_gemm_kernel(const unsigned short* __restrict__ Xb,
                     const unsigned short* __restrict__ Wb,
                     const int* __restrict__ counts,
                     const int* __restrict__ lists,
                     float* __restrict__ out) {
    __shared__ __align__(32) unsigned short As[TM][TK];
    __shared__ __align__(32) unsigned short Bs[TN][TK];

    int e   = blockIdx.z;
    int cnt = counts[e];
    int m0  = blockIdx.y * TM;
    if (m0 >= cnt) return;                 // tile beyond this expert's tokens
    int n0  = blockIdx.x * TN;
    const int* lst = lists + e * T_TOK;

    int tid = threadIdx.x;

    // cooperative-load addressing (gathered A rows, contiguous W rows)
    int a_row   = tid >> 2;                // 0..63
    int a_chunk = tid & 3;                 // 4 x 8 bf16 per row
    int a_tok   = lst[imin(m0 + a_row, cnt - 1)];
    const unsigned short* a_src = Xb + (size_t)a_tok * H_DIM + a_chunk * 8;

    int b_row  = tid >> 1;                 // 0..127
    int b_half = tid & 1;                  // 2 x 16 bf16 per row
    const unsigned short* b_src =
        Wb + (size_t)e * H_DIM * H_DIM + (size_t)(n0 + b_row) * H_DIM + b_half * 16;

    int wave = tid >> 5;
    int lane = tid & 31;
    int wm = wave & 1;                     // 2 wave-rows  (32 M each)
    int wn = wave >> 1;                    // 4 wave-cols  (32 N each)
    int lhi = lane >> 4;
    int llo = lane & 15;

    v8f zero = {};
    v8f c[2][2];
#pragma unroll
    for (int i = 0; i < 2; ++i)
#pragma unroll
        for (int j = 0; j < 2; ++j) c[i][j] = zero;

    for (int k = 0; k < H_DIM; k += TK) {
        uint4 av  = *(const uint4*)(a_src + k);
        uint4 bv0 = *(const uint4*)(b_src + k);
        uint4 bv1 = *(const uint4*)(b_src + k + 8);
        // speculative prefetch of next K tile (global_prefetch_b8)
        __builtin_prefetch(a_src + k + TK, 0, 1);
        __builtin_prefetch(b_src + k + TK, 0, 1);

        *(uint4*)(&As[a_row][a_chunk * 8])      = av;
        *(uint4*)(&Bs[b_row][b_half * 16])      = bv0;
        *(uint4*)(&Bs[b_row][b_half * 16 + 8])  = bv1;
        __syncthreads();

#pragma unroll
        for (int i = 0; i < 2; ++i) {
            int ar = wm * 32 + i * 16 + llo;
            V16u a;
            a.h.lo = *(const v8bf*)(&As[ar][lhi * 16]);
            a.h.hi = *(const v8bf*)(&As[ar][lhi * 16 + 8]);
#pragma unroll
            for (int j = 0; j < 2; ++j) {
                int br = wn * 32 + j * 16 + llo;
                v16bf b = *(const v16bf*)(&Bs[br][lhi * 16]);
                c[i][j] = __builtin_amdgcn_wmma_f32_16x16x32_bf16(
                    false, a.v, false, b, (short)0, c[i][j], false, false);
            }
        }
        __syncthreads();
    }

    // C/D layout: element r of v8f at lane L -> M = r + 8*(L>>4), N = L&15.
    // Each token appears under 2 experts -> accumulate with f32 atomics.
#pragma unroll
    for (int i = 0; i < 2; ++i) {
#pragma unroll
        for (int j = 0; j < 2; ++j) {
#pragma unroll
            for (int r = 0; r < 8; ++r) {
                int m = m0 + wm * 32 + i * 16 + r + 8 * lhi;
                if (m < cnt) {
                    int tok = lst[m];
                    int h = n0 + wn * 32 + j * 16 + llo;
                    unsafeAtomicAdd(&out[(size_t)tok * H_DIM + h], c[i][j][r]);
                }
            }
        }
    }
}

// ---------------------------------------------------------------------------
extern "C" void kernel_launch(void* const* d_in, const int* in_sizes, int n_in,
                              void* d_out, int out_size, void* d_ws, size_t ws_size,
                              hipStream_t stream) {
    const float* hs = (const float*)d_in[0];   // [2,1024,1024] hidden_states
    const float* rw = (const float*)d_in[1];   // [8,1024]      router_w
    const float* rb = (const float*)d_in[2];   // [8]           router_b
    const float* ew = (const float*)d_in[3];   // [8,1024,1024] expert_w
    const float* eb = (const float*)d_in[4];   // [8,1024]      expert_b
    float* out = (float*)d_out;                // [2,1024,1024]

    char* ws = (char*)d_ws;
    int* counts          = (int*)(ws + 0);                    //    32 B
    int* top2            = (int*)(ws + 1024);                 //   8 KB
    int* lists           = (int*)(ws + 16384);                //  64 KB
    unsigned short* Xb   = (unsigned short*)(ws + (1 << 17)); //   4 MB
    unsigned short* Wb   = (unsigned short*)(ws + (8u << 20));//  16 MB

    zero_counts_kernel<<<1, 32, 0, stream>>>(counts);
    cvt_bf16_kernel<<<2048, 256, 0, stream>>>(hs, Xb, T_TOK * H_DIM);
    cvt_bf16_kernel<<<8192, 256, 0, stream>>>(ew, Wb, N_EXP * H_DIM * H_DIM);
    router_kernel<<<T_TOK / 8, 256, 0, stream>>>(hs, rw, rb, counts, lists, top2);
    bias_init_kernel<<<T_TOK, 256, 0, stream>>>(eb, top2, out);

    dim3 grid(H_DIM / TN, T_TOK / TM, N_EXP);  // 8 x 32 x 8
    moe_gemm_kernel<<<grid, 256, 0, stream>>>(Xb, Wb, counts, lists, out);
}
